// PLEModel_68710886801904
// MI455X (gfx1250) — matrix operand order repaired and checked by reading, not compiled, not run on hardware
//
#include <hip/hip_runtime.h>

// ---------------------------------------------------------------------------
// Types for CDNA5 WMMA
// ---------------------------------------------------------------------------
typedef __attribute__((ext_vector_type(16))) __bf16        v16bf;
typedef __attribute__((ext_vector_type(8)))  float         v8f;
typedef __attribute__((ext_vector_type(4)))  unsigned int  v4u;

union FragBF { v4u q[2]; v16bf v; };

#define BN_INV 0.999995000037499687f  // 1/sqrt(1+1e-5), eval BatchNorm

__device__ __forceinline__ unsigned short f2bf(float f) {
  unsigned int u = __float_as_uint(f);
  u += 0x7FFFu + ((u >> 16) & 1u);               // round-to-nearest-even
  return (unsigned short)(u >> 16);
}
__device__ __forceinline__ float bf2f(unsigned short h) {
  return __uint_as_float(((unsigned int)h) << 16);
}

// ---------------------------------------------------------------------------
// bf16 GEMM with fused scale/offset (+ReLU) epilogue.
// C[M,N] = epilogue(A[M,K] * B[K,N]); A,B bf16 row-major, fp32 accumulate.
// REQUIRES: M % 64 == 0, N % BN == 0, K % 64 == 0, scl/off non-null.
// Block tile 64 x BN, 2*BN threads (BN/16 waves), wave tile 32x32 (2x2 WMMA).
// K tile depth 64 = two 32-deep WMMA sub-steps per LDS stage (8 WMMA/barrier).
// RELU / OUTBF16 are compile-time -> straight-line epilogue.
// ---------------------------------------------------------------------------
template <int BN, int RELU, int OUTBF16>
__global__ __launch_bounds__(2 * BN)
void gemm_bf16_kernel(const unsigned short* __restrict__ A,
                      const unsigned short* __restrict__ Bm,
                      void* __restrict__ Cout,
                      const float* __restrict__ scl,
                      const float* __restrict__ off,
                      int M, int N, int K)
{
  constexpr int NT  = 2 * BN;   // threads per block
  constexpr int LDA = 72;       // padded stride (ushorts): 144B rows -> 16B
  constexpr int LDB = 72;       //   aligned frag bases + spread LDS banks
  __shared__ __align__(16) unsigned short As[64 * LDA];
  __shared__ __align__(16) unsigned short Bs[BN * LDB];   // transposed: [n][k]

  const int tid  = threadIdx.x;
  const int wave = tid >> 5;
  const int lane = tid & 31;
  const int half = lane >> 4;   // wave32: lanes 0-15 / 16-31
  const int l16  = lane & 15;
  const int wm   = wave & 1;        // 2 wave-rows of 32
  const int wn   = wave >> 1;       // BN/32 wave-cols of 32
  const int mBase = blockIdx.y * 64;
  const int nBase = blockIdx.x * BN;

  v8f acc[2][2];
  #pragma unroll
  for (int i = 0; i < 2; ++i)
    #pragma unroll
    for (int j = 0; j < 2; ++j)
      #pragma unroll
      for (int r = 0; r < 8; ++r) acc[i][j][r] = 0.0f;

  const int kTiles = K >> 6;
  for (int kt = 0; kt < kTiles; ++kt) {
    // ---- stage A tile (64 x 64) into LDS, b128 chunks, no guards ----
    #pragma unroll
    for (int ch = 0; ch < 256 / BN; ++ch) {
      int cid = tid + ch * NT;
      int lin = cid << 3;
      int row = lin >> 6;
      int col = lin & 63;
      v4u val = *(const v4u*)&A[(size_t)(mBase + row) * K + (kt << 6) + col];
      *(v4u*)&As[row * LDA + col] = val;
    }
    // ---- stage B tile (64 x BN) into LDS transposed: Bs[n][k] ----
    #pragma unroll
    for (int ch = 0; ch < 4; ++ch) {
      int cid = tid + ch * NT;
      int lin = cid << 3;
      int k = lin / BN;
      int n = lin % BN;
      v4u v = *(const v4u*)&Bm[(size_t)((kt << 6) + k) * N + nBase + n];
      unsigned short tmp[8];
      tmp[0] = (unsigned short)(v.x & 0xffff); tmp[1] = (unsigned short)(v.x >> 16);
      tmp[2] = (unsigned short)(v.y & 0xffff); tmp[3] = (unsigned short)(v.y >> 16);
      tmp[4] = (unsigned short)(v.z & 0xffff); tmp[5] = (unsigned short)(v.z >> 16);
      tmp[6] = (unsigned short)(v.w & 0xffff); tmp[7] = (unsigned short)(v.w >> 16);
      #pragma unroll
      for (int e = 0; e < 8; ++e) Bs[(n + e) * LDB + k] = tmp[e];
    }
    __syncthreads();

    // ---- two 32-deep WMMA sub-steps over the staged tile ----
    #pragma unroll
    for (int ks = 0; ks < 2; ++ks) {
      // A (16x32, MxK): lane row = M, elems 0..7 -> K = 8*half + e,
      //                 elems 8..15 -> K = 16 + 8*half + (e-8)
      FragBF afr[2], bfr[2];
      #pragma unroll
      for (int sm = 0; sm < 2; ++sm) {
        int row = wm * 32 + sm * 16 + l16;
        const unsigned short* p = &As[row * LDA + ks * 32 + 8 * half];
        afr[sm].q[0] = *(const v4u*)p;
        afr[sm].q[1] = *(const v4u*)(p + 16);
      }
      // B (32x16, KxN): lane col = N, elems 0..7 -> K = 8*half + e, elems 8..15 -> K = 16 + 8*half + (e-8)
      #pragma unroll
      for (int sn = 0; sn < 2; ++sn) {
        int coln = wn * 32 + sn * 16 + l16;
        const unsigned short* p = &Bs[coln * LDB + ks * 32 + 8 * half];
        bfr[sn].q[0] = *(const v4u*)p;
        bfr[sn].q[1] = *(const v4u*)(p + 16);
      }
      #pragma unroll
      for (int sm = 0; sm < 2; ++sm)
        #pragma unroll
        for (int sn = 0; sn < 2; ++sn)
          acc[sm][sn] = __builtin_amdgcn_wmma_f32_16x16x32_bf16(
              false, afr[sm].v, false, bfr[sn].v, (short)0, acc[sm][sn],
              false, false);
    }
    __syncthreads();
  }

  // ---- epilogue: val = acc*s[n] + c[n], optional ReLU, bf16 or f32 store ----
  #pragma unroll
  for (int sm = 0; sm < 2; ++sm) {
    #pragma unroll
    for (int sn = 0; sn < 2; ++sn) {
      int gcol = nBase + wn * 32 + sn * 16 + l16;
      float s = scl[gcol];
      float c = off[gcol];
      #pragma unroll
      for (int r = 0; r < 8; ++r) {
        int grow = mBase + wm * 32 + sm * 16 + 8 * half + r;  // D: M = r + 8*half
        float v = acc[sm][sn][r] * s + c;
        if (RELU) v = fmaxf(v, 0.0f);
        if (OUTBF16) ((unsigned short*)Cout)[(size_t)grow * N + gcol] = f2bf(v);
        else         ((float*)Cout)[(size_t)grow * N + gcol] = v;
      }
    }
  }
}

// ---------------------------------------------------------------------------
// Elementwise / packing helpers
// ---------------------------------------------------------------------------
__global__ void cvt_f32_bf16_kernel(const float* __restrict__ s,
                                    unsigned short* __restrict__ d, int n) {
  int i = blockIdx.x * blockDim.x + threadIdx.x;
  if (i < n) d[i] = f2bf(s[i]);
}

__global__ void ones_kernel(float* __restrict__ d, int n) {
  int i = blockIdx.x * blockDim.x + threadIdx.x;
  if (i < n) d[i] = 1.0f;
}

// s = INV*gamma, c = bias*s + beta   (folds Linear bias + eval-BN)
__global__ void make_sc_kernel(const float* __restrict__ g, const float* __restrict__ be,
                               const float* __restrict__ b, float* __restrict__ s,
                               float* __restrict__ c, int n) {
  int i = blockIdx.x * blockDim.x + threadIdx.x;
  if (i < n) { float sv = BN_INV * g[i]; s[i] = sv; c[i] = b[i] * sv + be[i]; }
}

// W0 pack: [2176, 24*512] bf16, expert order [t0e0..t0e7, t1e0..t1e7, sh0..sh7]
__global__ void pack_w0_kernel(const float* __restrict__ te0W,
                               const float* __restrict__ se0W,
                               unsigned short* __restrict__ dst) {
  int idx = blockIdx.x * blockDim.x + threadIdx.x;
  if (idx >= 2176 * 12288) return;
  int i = idx / 12288, n = idx % 12288;
  float v;
  if (n < 8192) {
    int t = n >> 12, e = (n >> 9) & 7, o = n & 511;
    v = te0W[(((size_t)(t * 8 + e)) * 2176 + i) * 512 + o];
  } else {
    int m = n - 8192; int e = m >> 9, o = m & 511;
    v = se0W[(((size_t)e) * 2176 + i) * 512 + o];
  }
  dst[idx] = f2bf(v);
}

// W1 pack: 3 blocks [512, 8*256] bf16 (t0, t1, shared)
__global__ void pack_w1_kernel(const float* __restrict__ te1W,
                               const float* __restrict__ se1W,
                               unsigned short* __restrict__ dst) {
  int idx = blockIdx.x * blockDim.x + threadIdx.x;
  if (idx >= 3 * 512 * 2048) return;
  int blk = idx / (512 * 2048);
  int r = idx % (512 * 2048);
  int i = r / 2048, n = r % 2048;
  int e = n >> 8, o = n & 255;
  float v = (blk < 2)
      ? te1W[(((size_t)(blk * 8 + e)) * 512 + i) * 256 + o]
      : se1W[(((size_t)e) * 512 + i) * 256 + o];
  dst[idx] = f2bf(v);
}

// Gate0 weights: [2176, 64] (cols 0-15 t0, 16-31 t1, 32-55 shared gate, pad 0)
__global__ void pack_g0w_kernel(const float* __restrict__ tg0W,
                                const float* __restrict__ sg0W,
                                unsigned short* __restrict__ dst) {
  int idx = blockIdx.x * blockDim.x + threadIdx.x;
  if (idx >= 2176 * 64) return;
  int i = idx >> 6, n = idx & 63;
  float v = 0.0f;
  if (n < 32)      { int t = n >> 4; v = tg0W[((size_t)t * 2176 + i) * 16 + (n & 15)]; }
  else if (n < 56) { v = sg0W[(size_t)i * 24 + (n - 32)]; }
  dst[idx] = f2bf(v);
}
__global__ void pack_cg0_kernel(const float* __restrict__ tg0b,
                                const float* __restrict__ sg0b,
                                float* __restrict__ cg) {
  int n = threadIdx.x;  // 64
  float v = 0.0f;
  if (n < 32) v = tg0b[n];
  else if (n < 56) v = sg0b[n - 32];
  cg[n] = v;
}

// Gate1 weights: per task [512, 64] (cols 0-15 valid)
__global__ void pack_g1w_kernel(const float* __restrict__ tg1W,
                                unsigned short* __restrict__ dst) {
  int idx = blockIdx.x * blockDim.x + threadIdx.x;
  if (idx >= 2 * 512 * 64) return;
  int t = idx / (512 * 64);
  int r = idx % (512 * 64);
  int i = r >> 6, n = r & 63;
  float v = (n < 16) ? tg1W[((size_t)t * 512 + i) * 16 + n] : 0.0f;
  dst[idx] = f2bf(v);
}
__global__ void pack_cg1_kernel(const float* __restrict__ tg1b, float* __restrict__ cg) {
  int idx = threadIdx.x;  // 128
  int t = idx >> 6, n = idx & 63;
  cg[idx] = (n < 16) ? tg1b[t * 16 + n] : 0.0f;
}

// ---------------------------------------------------------------------------
// Embedding gather + numeric dense -> X [4096, 2176] bf16
// ---------------------------------------------------------------------------
__global__ void embed_kernel(const int* __restrict__ cat, const float* __restrict__ numx,
                             const float* __restrict__ embT, const float* __restrict__ numW,
                             const float* __restrict__ numb, unsigned short* __restrict__ X) {
  int idx = blockIdx.x * blockDim.x + threadIdx.x;
  if (idx >= 4096 * 2176) return;
  int b = idx / 2176, j = idx % 2176;
  float v;
  if (j < 2048) {
    int f = j >> 7, d = j & 127;
    long long row = (long long)cat[b * 16 + f] + (long long)f * 50000;
    v = embT[row * 128 + d];
  } else {
    int d = j - 2048;
    float a = numb[d];
    for (int k = 0; k < 63; ++k) a += numx[b * 63 + k] * numW[k * 128 + d];
    v = a;
  }
  X[idx] = f2bf(v);
}

// ---------------------------------------------------------------------------
// Softmax over gate segments (in-place on fp32 gate logits)
// ---------------------------------------------------------------------------
__global__ void softmax0_kernel(float* __restrict__ G0) {
  int idx = blockIdx.x * blockDim.x + threadIdx.x;
  if (idx >= 3 * 4096) return;
  int kind = idx / 4096, b = idx % 4096;
  float* row = G0 + (size_t)b * 64 + (kind == 0 ? 0 : (kind == 1 ? 16 : 32));
  int len = (kind == 2) ? 24 : 16;
  float mx = row[0];
  for (int i = 1; i < len; ++i) mx = fmaxf(mx, row[i]);
  float s = 0.0f;
  for (int i = 0; i < len; ++i) { float e = __expf(row[i] - mx); row[i] = e; s += e; }
  float inv = 1.0f / s;
  for (int i = 0; i < len; ++i) row[i] *= inv;
}
__global__ void softmax1_kernel(float* __restrict__ G1) {
  int idx = blockIdx.x * blockDim.x + threadIdx.x;
  if (idx >= 2 * 4096) return;
  float* row = G1 + (size_t)idx * 64;
  float mx = row[0];
  for (int i = 1; i < 16; ++i) mx = fmaxf(mx, row[i]);
  float s = 0.0f;
  for (int i = 0; i < 16; ++i) { float e = __expf(row[i] - mx); row[i] = e; s += e; }
  float inv = 1.0f / s;
  for (int i = 0; i < 16; ++i) row[i] *= inv;
}

// ---------------------------------------------------------------------------
// Gate-weighted expert mixing
// ---------------------------------------------------------------------------
__global__ void mixing0_kernel(const float* __restrict__ G0,
                               const unsigned short* __restrict__ E0,
                               unsigned short* __restrict__ X1) {
  int idx = blockIdx.x * blockDim.x + threadIdx.x;
  if (idx >= 3 * 4096 * 512) return;
  int blk = idx / (4096 * 512);
  int r = idx % (4096 * 512);
  int b = r >> 9, o = r & 511;
  const unsigned short* eb = E0 + (size_t)b * 12288;
  float acc = 0.0f;
  if (blk < 2) {            // task gates over [own 8 experts ++ 8 shared]
    const float* g = G0 + (size_t)b * 64 + blk * 16;
    #pragma unroll
    for (int gi = 0; gi < 16; ++gi) {
      int ex = (gi < 8) ? (blk * 8 + gi) : (8 + gi);   // shared base = 16
      acc += g[gi] * bf2f(eb[ex * 512 + o]);
    }
  } else {                  // shared gate over all 24 experts (task-major ++ shared)
    const float* g = G0 + (size_t)b * 64 + 32;
    #pragma unroll
    for (int gi = 0; gi < 24; ++gi) acc += g[gi] * bf2f(eb[gi * 512 + o]);
  }
  X1[idx] = f2bf(acc);
}

__global__ void mixing1_kernel(const float* __restrict__ G1,
                               const unsigned short* __restrict__ E1,
                               unsigned short* __restrict__ X2) {
  int idx = blockIdx.x * blockDim.x + threadIdx.x;
  if (idx >= 2 * 4096 * 256) return;
  int t = idx / (4096 * 256);
  int r = idx % (4096 * 256);
  int b = r >> 8, o = r & 255;
  const float* g = G1 + ((size_t)t * 4096 + b) * 64;
  const unsigned short* et = E1 + ((size_t)t * 4096 + b) * 2048;
  const unsigned short* es = E1 + (size_t)2 * 4096 * 2048 + (size_t)b * 2048;
  float acc = 0.0f;
  #pragma unroll
  for (int gi = 0; gi < 8; ++gi)  acc += g[gi] * bf2f(et[gi * 256 + o]);
  #pragma unroll
  for (int gi = 8; gi < 16; ++gi) acc += g[gi] * bf2f(es[(gi - 8) * 256 + o]);
  X2[idx] = f2bf(acc);
}

// ---------------------------------------------------------------------------
// Final 64-wide dot + sigmoid -> out[T,B]
// ---------------------------------------------------------------------------
__global__ void final_kernel(const unsigned short* __restrict__ H2,
                             const float* __restrict__ twoW,
                             const float* __restrict__ twob,
                             float* __restrict__ out) {
  int idx = blockIdx.x * blockDim.x + threadIdx.x;
  if (idx >= 8192) return;
  int t = idx >> 12, b = idx & 4095;
  const unsigned short* h = H2 + ((size_t)t * 4096 + b) * 64;
  float acc = twob[t];
  #pragma unroll
  for (int i = 0; i < 64; ++i) acc += bf2f(h[i]) * twoW[t * 64 + i];
  out[idx] = 1.0f / (1.0f + __expf(-acc));
}

// ---------------------------------------------------------------------------
// Launcher
// ---------------------------------------------------------------------------
#define GRID1(n) dim3((unsigned)(((n) + 255) / 256))

extern "C" void kernel_launch(void* const* d_in, const int* in_sizes, int n_in,
                              void* d_out, int out_size, void* d_ws, size_t ws_size,
                              hipStream_t stream) {
  (void)in_sizes; (void)n_in; (void)out_size; (void)ws_size;
  const int*   cat  = (const int*)d_in[0];
  const float* numx = (const float*)d_in[1];
  const float* embT = (const float*)d_in[2];
  const float* numW = (const float*)d_in[3];
  const float* numb = (const float*)d_in[4];
  const float* se0W = (const float*)d_in[5],  *se0b = (const float*)d_in[6];
  const float* se0g = (const float*)d_in[7],  *se0be = (const float*)d_in[8];
  const float* te0W = (const float*)d_in[9],  *te0b = (const float*)d_in[10];
  const float* te0g = (const float*)d_in[11], *te0be = (const float*)d_in[12];
  const float* tg0W = (const float*)d_in[13], *tg0b = (const float*)d_in[14];
  const float* sg0W = (const float*)d_in[15], *sg0b = (const float*)d_in[16];
  const float* se1W = (const float*)d_in[17], *se1b = (const float*)d_in[18];
  const float* se1g = (const float*)d_in[19], *se1be = (const float*)d_in[20];
  const float* te1W = (const float*)d_in[21], *te1b = (const float*)d_in[22];
  const float* te1g = (const float*)d_in[23], *te1be = (const float*)d_in[24];
  const float* tg1W = (const float*)d_in[25], *tg1b = (const float*)d_in[26];
  const float* tw0W = (const float*)d_in[27], *tw0b = (const float*)d_in[28];
  const float* tw0g = (const float*)d_in[29], *tw0be = (const float*)d_in[30];
  const float* tw1W = (const float*)d_in[31], *tw1b = (const float*)d_in[32];
  const float* tw1g = (const float*)d_in[33], *tw1be = (const float*)d_in[34];
  const float* twoW = (const float*)d_in[35], *twob = (const float*)d_in[36];

  char* ws = (char*)d_ws;
  size_t bump = 0;
  auto alloc = [&](size_t bytes) -> void* {
    void* p = ws + bump;
    bump += (bytes + 255) & ~(size_t)255;
    return p;
  };
  unsigned short* Xemb = (unsigned short*)alloc((size_t)4096 * 2176 * 2);
  unsigned short* W0   = (unsigned short*)alloc((size_t)2176 * 12288 * 2);
  float* S0  = (float*)alloc(12288 * 4);
  float* C0  = (float*)alloc(12288 * 4);
  unsigned short* G0W = (unsigned short*)alloc((size_t)2176 * 64 * 2);
  float* CG0 = (float*)alloc(64 * 4);
  unsigned short* E0  = (unsigned short*)alloc((size_t)4096 * 12288 * 2);
  float* G0  = (float*)alloc((size_t)4096 * 64 * 4);
  unsigned short* X1  = (unsigned short*)alloc((size_t)3 * 4096 * 512 * 2);
  unsigned short* W1  = (unsigned short*)alloc((size_t)3 * 512 * 2048 * 2);
  float* S1  = (float*)alloc(3 * 2048 * 4);
  float* C1  = (float*)alloc(3 * 2048 * 4);
  unsigned short* G1W = (unsigned short*)alloc((size_t)2 * 512 * 64 * 2);
  float* CG1 = (float*)alloc(128 * 4);
  unsigned short* E1  = (unsigned short*)alloc((size_t)3 * 4096 * 2048 * 2);
  float* G1  = (float*)alloc((size_t)2 * 4096 * 64 * 4);
  unsigned short* X2  = (unsigned short*)alloc((size_t)2 * 4096 * 256 * 2);
  unsigned short* TW0 = (unsigned short*)alloc((size_t)2 * 256 * 128 * 2);
  float* TS0 = (float*)alloc(256 * 4);
  float* TC0 = (float*)alloc(256 * 4);
  unsigned short* TW1 = (unsigned short*)alloc((size_t)2 * 128 * 64 * 2);
  float* TS1 = (float*)alloc(128 * 4);
  float* TC1 = (float*)alloc(128 * 4);
  unsigned short* H1  = (unsigned short*)alloc((size_t)2 * 4096 * 128 * 2);
  unsigned short* H2  = (unsigned short*)alloc((size_t)2 * 4096 * 64 * 2);
  float* ONES = (float*)alloc(64 * 4);   // identity scale for gate GEMMs

  // ---- weight packing / epilogue folding ----
  ones_kernel<<<1, 64, 0, stream>>>(ONES, 64);
  pack_w0_kernel<<<GRID1(2176 * 12288), 256, 0, stream>>>(te0W, se0W, W0);
  make_sc_kernel<<<GRID1(8192), 256, 0, stream>>>(te0g, te0be, te0b, S0, C0, 8192);
  make_sc_kernel<<<GRID1(4096), 256, 0, stream>>>(se0g, se0be, se0b, S0 + 8192, C0 + 8192, 4096);
  pack_g0w_kernel<<<GRID1(2176 * 64), 256, 0, stream>>>(tg0W, sg0W, G0W);
  pack_cg0_kernel<<<1, 64, 0, stream>>>(tg0b, sg0b, CG0);
  pack_w1_kernel<<<GRID1(3 * 512 * 2048), 256, 0, stream>>>(te1W, se1W, W1);
  make_sc_kernel<<<GRID1(4096), 256, 0, stream>>>(te1g, te1be, te1b, S1, C1, 4096);
  make_sc_kernel<<<GRID1(2048), 256, 0, stream>>>(se1g, se1be, se1b, S1 + 4096, C1 + 4096, 2048);
  pack_g1w_kernel<<<GRID1(2 * 512 * 64), 256, 0, stream>>>(tg1W, G1W);
  pack_cg1_kernel<<<1, 128, 0, stream>>>(tg1b, CG1);
  cvt_f32_bf16_kernel<<<GRID1(2 * 256 * 128), 256, 0, stream>>>(tw0W, TW0, 2 * 256 * 128);
  make_sc_kernel<<<GRID1(256), 256, 0, stream>>>(tw0g, tw0be, tw0b, TS0, TC0, 256);
  cvt_f32_bf16_kernel<<<GRID1(2 * 128 * 64), 256, 0, stream>>>(tw1W, TW1, 2 * 128 * 64);
  make_sc_kernel<<<GRID1(128), 256, 0, stream>>>(tw1g, tw1be, tw1b, TS1, TC1, 128);

  // ---- embedding + numeric ----
  embed_kernel<<<GRID1(4096 * 2176), 256, 0, stream>>>(cat, numx, embT, numW, numb, Xemb);

  // ---- PLE layer 0: one fused 24-expert GEMM + gate GEMM ----
  gemm_bf16_kernel<128, 1, 1><<<dim3(12288 / 128, 4096 / 64), 256, 0, stream>>>(
      Xemb, W0, E0, S0, C0, 4096, 12288, 2176);
  gemm_bf16_kernel<64, 0, 0><<<dim3(1, 4096 / 64), 128, 0, stream>>>(
      Xemb, G0W, G0, ONES, CG0, 4096, 64, 2176);
  softmax0_kernel<<<GRID1(3 * 4096), 256, 0, stream>>>(G0);
  mixing0_kernel<<<GRID1(3 * 4096 * 512), 256, 0, stream>>>(G0, E0, X1);

  // ---- PLE layer 1: 3 grouped expert GEMMs + 2 gate GEMMs ----
  for (int blk = 0; blk < 3; ++blk)
    gemm_bf16_kernel<128, 1, 1><<<dim3(2048 / 128, 4096 / 64), 256, 0, stream>>>(
        X1 + (size_t)blk * 4096 * 512, W1 + (size_t)blk * 512 * 2048,
        E1 + (size_t)blk * 4096 * 2048, S1 + blk * 2048, C1 + blk * 2048,
        4096, 2048, 512);
  for (int t = 0; t < 2; ++t)
    gemm_bf16_kernel<64, 0, 0><<<dim3(1, 4096 / 64), 128, 0, stream>>>(
        X1 + (size_t)t * 4096 * 512, G1W + (size_t)t * 512 * 64,
        G1 + (size_t)t * 4096 * 64, ONES, CG1 + t * 64,
        4096, 64, 512);
  softmax1_kernel<<<GRID1(2 * 4096), 256, 0, stream>>>(G1);
  mixing1_kernel<<<GRID1(2 * 4096 * 256), 256, 0, stream>>>(G1, E1, X2);

  // ---- towers ----
  for (int t = 0; t < 2; ++t)
    gemm_bf16_kernel<128, 1, 1><<<dim3(1, 4096 / 64), 256, 0, stream>>>(
        X2 + (size_t)t * 4096 * 256, TW0 + (size_t)t * 256 * 128,
        H1 + (size_t)t * 4096 * 128, TS0 + t * 128, TC0 + t * 128,
        4096, 128, 256);
  for (int t = 0; t < 2; ++t)
    gemm_bf16_kernel<64, 1, 1><<<dim3(1, 4096 / 64), 128, 0, stream>>>(
        H1 + (size_t)t * 4096 * 128, TW1 + (size_t)t * 128 * 64,
        H2 + (size_t)t * 4096 * 64, TS1 + t * 64, TC1 + t * 64,
        4096, 64, 128);
  final_kernel<<<GRID1(8192), 256, 0, stream>>>(H2, twoW, twob, (float*)d_out);
}
